// NeuralLogicProgramming_24404004176248
// MI455X (gfx1250) — compile-verified, matrix operand order checked by reading, not run on hardware
//
#include <hip/hip_runtime.h>

#define N_NODES   20000
#define N1        20001
#define N_EDGES   1000000
#define N_STEP    6
#define HID       128
#define NRELX2    24

typedef __attribute__((ext_vector_type(16))) _Float16 v16h;
typedef __attribute__((ext_vector_type(8)))  float    v8f;

__device__ __forceinline__ float sigf(float x) { return 1.0f / (1.0f + __expf(-x)); }

// 16-bit A/B WMMA fragment swizzle: k_local (0..31) -> lane group g and slot.
__device__ __forceinline__ int frag_g(int kl)    { return (kl >> 3) & 1; }
__device__ __forceinline__ int frag_slot(int kl) {
    int v = ((kl >> 4) << 2) + ((kl & 7) >> 1);
    return 2 * v + (kl & 1);
}

// ---------------------------------------------------------------------------
// Phase A: LSTM (WMMA f16->f32) + attention softmaxes + edge-type weight
// softmaxes. One block, 8 wave32s. Gates GEMM: [x|h](64x256) @ W(256x512),
// 16x16x32 tiles; cell state stays in accumulator-layout registers across all
// 6 timesteps. A (LDS) and B (global half) are kept in fragment-swizzled
// layout so each fragment is two 128-bit loads.
// ---------------------------------------------------------------------------
__global__ __launch_bounds__(256, 1)
void lstm_attn_kernel(const float* __restrict__ query_emb,
                      const float* __restrict__ w_ih, const float* __restrict__ w_hh,
                      const float* __restrict__ b_ih, const float* __restrict__ b_hh,
                      const float* __restrict__ wl_w, const float* __restrict__ wl_b,
                      const int*   __restrict__ r_set,
                      float* __restrict__ hiddenWS, float* __restrict__ attnWS,
                      float* __restrict__ weightWS, _Float16* __restrict__ Wf)
{
    // A fragments: AhS[((mT*8 + kT)*32 + lane)*16 + slot]; 4*8 frags, 32KB
    __shared__ _Float16 AhS[4 * 8 * 32 * 16];
    const int tid  = threadIdx.x;
    const int lane = tid & 31;
    const int wave = tid >> 5;

    // B fragments, fragment-swizzled: Wf[((kT*32 + nT)*32 + lane)*16 + slot]
    // element (k,j): k<128 -> w_ih[j][k], k>=128 -> w_hh[j][k-128]
    for (int idx = tid; idx < 256 * 512; idx += 256) {
        int k = idx >> 9, j = idx & 511;
        float w = (k < 128) ? w_ih[j * 128 + k] : w_hh[j * 128 + (k - 128)];
        int kT = k >> 5, kl = k & 31, nT = j >> 4;
        int fl = frag_g(kl) * 16 + (j & 15);
        Wf[((size_t)(kT * 32 + nT) * 32 + fl) * 16 + frag_slot(kl)] = (_Float16)w;
    }
    // zero h-part of A (k = 128..255 -> kT 4..7)
    for (int idx = tid; idx < 64 * 128; idx += 256) {
        int b = idx >> 7, k = 128 + (idx & 127);
        int kT = k >> 5, kl = k & 31;
        int fl = frag_g(kl) * 16 + (b & 15);
        AhS[(((b >> 4) * 8 + kT) * 32 + fl) * 16 + frag_slot(kl)] = (_Float16)0.0f;
    }

    float cstate[4][8];
#pragma unroll
    for (int p = 0; p < 4; ++p)
#pragma unroll
        for (int r = 0; r < 8; ++r) cstate[p][r] = 0.0f;

    for (int s = 0; s < N_STEP; ++s) {
        // stage x = query_emb[q_index[s]] into A (kT 0..3)
        for (int idx = tid; idx < 64 * 128; idx += 256) {
            int b = idx >> 7, k = idx & 127;
            int q = (s < N_STEP - 1) ? r_set[b] : 12;     // last step: NUM_REL
            int kT = k >> 5, kl = k & 31;
            int fl = frag_g(kl) * 16 + (b & 15);
            AhS[(((b >> 4) * 8 + kT) * 32 + fl) * 16 + frag_slot(kl)] =
                (_Float16)query_emb[q * HID + k];
        }
        __syncthreads();

        float hout[4][8];
#pragma unroll
        for (int p = 0; p < 4; ++p) {
            int P  = wave * 4 + p;                 // 32 tiles: 4 mTiles x 8 nTiles
            int mT = P >> 3, nT = P & 7;
            v8f aI = {}, aF = {}, aG = {}, aO = {};
            for (int kT = 0; kT < 8; ++kT) {
                v16h a  = *(const v16h*)&AhS[((mT * 8 + kT) * 32 + lane) * 16];
                v16h bi = *(const v16h*)&Wf[((size_t)(kT * 32 + nT)      * 32 + lane) * 16];
                v16h bf = *(const v16h*)&Wf[((size_t)(kT * 32 + nT +  8) * 32 + lane) * 16];
                v16h bg = *(const v16h*)&Wf[((size_t)(kT * 32 + nT + 16) * 32 + lane) * 16];
                v16h bo = *(const v16h*)&Wf[((size_t)(kT * 32 + nT + 24) * 32 + lane) * 16];
                aI = __builtin_amdgcn_wmma_f32_16x16x32_f16(false, a, false, bi, (short)0, aI, false, false);
                aF = __builtin_amdgcn_wmma_f32_16x16x32_f16(false, a, false, bf, (short)0, aF, false, false);
                aG = __builtin_amdgcn_wmma_f32_16x16x32_f16(false, a, false, bg, (short)0, aG, false, false);
                aO = __builtin_amdgcn_wmma_f32_16x16x32_f16(false, a, false, bo, (short)0, aO, false, false);
            }
            const int nD = nT * 16 + (lane & 15);  // D layout: lane -> N
            float biasI = b_ih[nD]       + b_hh[nD];
            float biasF = b_ih[128 + nD] + b_hh[128 + nD];
            float biasG = b_ih[256 + nD] + b_hh[256 + nD];
            float biasO = b_ih[384 + nD] + b_hh[384 + nD];
#pragma unroll
            for (int r = 0; r < 8; ++r) {          // D layout: reg r -> M (+8 if lane>=16)
                float ig = sigf(aI[r] + biasI);
                float fg = sigf(aF[r] + biasF);
                float gg = tanhf(aG[r] + biasG);
                float og = sigf(aO[r] + biasO);
                float c  = fg * cstate[p][r] + ig * gg;
                cstate[p][r] = c;
                hout[p][r]   = og * tanhf(c);
            }
        }
        __syncthreads();   // all A reads complete before h overwrite
#pragma unroll
        for (int p = 0; p < 4; ++p) {
            int P  = wave * 4 + p;
            int mT = P >> 3, nT = P & 7;
            int nD = nT * 16 + (lane & 15);
            int k  = 128 + nD, kT = k >> 5, kl = k & 31;
            int fg_ = frag_g(kl) * 16, fs = frag_slot(kl);
#pragma unroll
            for (int r = 0; r < 8; ++r) {
                int bIdx = mT * 16 + r + (lane >> 4) * 8;
                float h  = hout[p][r];
                hiddenWS[((size_t)s * 64 + bIdx) * HID + nD] = h;
                AhS[(((bIdx >> 4) * 8 + kT) * 32 + fg_ + (bIdx & 15)) * 16 + fs] = (_Float16)h;
            }
        }
        // next iteration's x-fill writes disjoint region; its barrier orders all
    }
    __syncthreads();

    // Precompute attn[i][t][b] and weight[i][e][b] for every step.
    for (int idx = tid; idx < N_STEP * 64; idx += 256) {
        int i = idx >> 6, b = idx & 63;
        const float* hi = &hiddenWS[((size_t)i * 64 + b) * HID];

        float lg[N_STEP];
        float mx = -1e30f;
        for (int t = 0; t <= i; ++t) {
            const float* ht = &hiddenWS[((size_t)t * 64 + b) * HID];
            float d = 0.0f;
            for (int k = 0; k < HID; ++k) d += hi[k] * ht[k];
            lg[t] = d; mx = fmaxf(mx, d);
        }
        float sum = 0.0f;
        for (int t = 0; t <= i; ++t) { lg[t] = __expf(lg[t] - mx); sum += lg[t]; }
        float inv = 1.0f / sum;
        for (int t = 0; t <= i; ++t)
            attnWS[((size_t)i * N_STEP + t) * 64 + b] = lg[t] * inv;

        float wl[NRELX2];
        mx = -1e30f;
        for (int e = 0; e < NRELX2; ++e) {
            float d = wl_b[e];
            for (int k = 0; k < HID; ++k) d += hi[k] * wl_w[e * HID + k];
            wl[e] = d; mx = fmaxf(mx, d);
        }
        sum = 0.0f;
        for (int e = 0; e < NRELX2; ++e) { wl[e] = __expf(wl[e] - mx); sum += wl[e]; }
        inv = 1.0f / sum;
        for (int e = 0; e < NRELX2; ++e)
            weightWS[((size_t)i * NRELX2 + e) * 64 + b] = wl[e] * inv;
    }
}

// ---------------------------------------------------------------------------
// CSR build (once per call): degree histogram -> scan -> fill (src|et packed)
// ---------------------------------------------------------------------------
__global__ void deg_init_kernel(int* __restrict__ deg)
{
    int n = blockIdx.x * blockDim.x + threadIdx.x;
    if (n < N1) deg[n] = 0;
}

__global__ void deg_count_kernel(const int* __restrict__ node_out, int* __restrict__ deg)
{
    int e = blockIdx.x * blockDim.x + threadIdx.x;
    if (e < N_EDGES) atomicAdd(&deg[node_out[e]], 1);
}

__global__ __launch_bounds__(256, 1)
void scan_kernel(const int* __restrict__ deg, int* __restrict__ row_ptr, int* __restrict__ cursor)
{
    __shared__ int partial[256];
    const int CH = (N1 + 255) / 256;     // 79 nodes per thread
    int t = threadIdx.x;
    int start = t * CH;
    int end   = min(start + CH, N1);
    int s = 0;
    for (int n = start; n < end; ++n) s += deg[n];
    partial[t] = s;
    __syncthreads();
    if (t == 0) {
        int run = 0;
        for (int i = 0; i < 256; ++i) { int v = partial[i]; partial[i] = run; run += v; }
    }
    __syncthreads();
    int run = partial[t];
    for (int n = start; n < end; ++n) {
        row_ptr[n] = run; cursor[n] = run; run += deg[n];
    }
    if (end == N1 && start <= N1) row_ptr[N1] = run;   // total = N_EDGES
}

__global__ void fill_kernel(const int* __restrict__ node_in, const int* __restrict__ node_out,
                            const int* __restrict__ edge_type,
                            int* __restrict__ cursor, int* __restrict__ csr)
{
    int e = blockIdx.x * blockDim.x + threadIdx.x;
    if (e >= N_EDGES) return;
    int slot = atomicAdd(&cursor[node_out[e]], 1);
    csr[slot] = (node_in[e] & 0xFFFF) | (edge_type[e] << 16);   // src < 2^16, et < 24
}

// ---------------------------------------------------------------------------
// Phase B kernels (per step)
// ---------------------------------------------------------------------------
__global__ void inp_kernel(const float* __restrict__ attn,      // attn[t*64+b] for this step
                           const float* __restrict__ outbufs,   // prior steps' outputs
                           const int*   __restrict__ h_set,
                           float* __restrict__ inp, float* __restrict__ colsum, int stepi)
{
    int tid = blockIdx.x * blockDim.x + threadIdx.x;
    if (tid < 64) colsum[tid] = 0.0f;
    if (tid >= N1 * 64) return;
    int n = tid >> 6, b = tid & 63;
    float acc = 0.0f;
    for (int t = 1; t <= stepi; ++t)
        acc += attn[t * 64 + b] * outbufs[(size_t)(t - 1) * N1 * 64 + tid];
    if (n == h_set[b]) acc += attn[b];               // one-hot memory[0]
    inp[tid] = acc;
}

// One 64-thread group per destination node; register accumulation, no atomics.
__global__ void gather_kernel(const int* __restrict__ row_ptr, const int* __restrict__ csr,
                              const float* __restrict__ inp, const float* __restrict__ weight,
                              float* __restrict__ out)
{
    int n = blockIdx.x * 4 + (threadIdx.x >> 6);     // 4 nodes per 256-thread block
    int b = threadIdx.x & 63;
    if (n >= N1) return;
    int s0 = row_ptr[n], s1 = row_ptr[n + 1];
    float acc = 0.0f;
    for (int s = s0; s < s1; ++s) {
        int pk  = csr[s];                            // uniform per group -> broadcast
        int src = pk & 0xFFFF;
        int et  = pk >> 16;
        acc += inp[(size_t)src * 64 + b] * weight[et * 64 + b];
    }
    out[(size_t)n * 64 + b] = acc;
}

__global__ void colsum_kernel(const float* __restrict__ out, float* __restrict__ colsum)
{
    int b = threadIdx.x & 63;
    int n = blockIdx.x * 4 + (threadIdx.x >> 6);
    float acc = 0.0f;
    for (; n < N1; n += gridDim.x * 4) acc += out[(size_t)n * 64 + b];
    atomicAdd(&colsum[b], acc);
}

__global__ void norm_kernel(float* __restrict__ out, const float* __restrict__ colsum)
{
    int tid = blockIdx.x * blockDim.x + threadIdx.x;
    if (tid >= N1 * 64) return;
    out[tid] /= fmaxf(colsum[tid & 63], 1e-10f);
}

__global__ void score_kernel(const float* __restrict__ out5,
                             const int* __restrict__ t_index, const int* __restrict__ hr_inverse,
                             const float* __restrict__ lin_w, const float* __restrict__ lin_b,
                             float* __restrict__ dst)
{
    int tid = blockIdx.x * blockDim.x + threadIdx.x;
    if (tid >= 64 * 32) return;
    int t = t_index[tid], hr = hr_inverse[tid];
    dst[tid] = out5[(size_t)t * 64 + hr] * lin_w[0] + lin_b[0];
}

// ---------------------------------------------------------------------------
extern "C" void kernel_launch(void* const* d_in, const int* in_sizes, int n_in,
                              void* d_out, int out_size, void* d_ws, size_t ws_size,
                              hipStream_t stream)
{
    const float* query_emb  = (const float*)d_in[0];
    const float* w_ih       = (const float*)d_in[1];
    const float* w_hh       = (const float*)d_in[2];
    const float* b_ih       = (const float*)d_in[3];
    const float* b_hh       = (const float*)d_in[4];
    const float* wl_w       = (const float*)d_in[5];
    const float* wl_b       = (const float*)d_in[6];
    const float* lin_w      = (const float*)d_in[7];
    const float* lin_b      = (const float*)d_in[8];
    const int*   edge_index = (const int*)d_in[9];
    const int*   edge_type  = (const int*)d_in[10];
    const int*   h_set      = (const int*)d_in[11];
    const int*   r_set      = (const int*)d_in[12];
    const int*   t_index    = (const int*)d_in[13];
    const int*   hr_inverse = (const int*)d_in[14];
    const int*   node_in    = edge_index;
    const int*   node_out   = edge_index + N_EDGES;

    float* ws = (float*)d_ws;
    size_t off = 0;
    float*    hiddenWS = ws + off; off += (size_t)N_STEP * 64 * HID;     // 49152
    float*    attnWS   = ws + off; off += (size_t)N_STEP * N_STEP * 64;  // 2304
    float*    weightWS = ws + off; off += (size_t)N_STEP * NRELX2 * 64;  // 9216
    float*    colsum   = ws + off; off += 64;
    _Float16* Wf       = (_Float16*)(ws + off); off += (256 * 512) / 2;  // 65536 floats
    float*    inp      = ws + off; off += (size_t)N1 * 64;               // 1,280,064
    float*    outbufs  = ws + off; off += (size_t)N_STEP * N1 * 64;      // 7,680,384
    int*      deg      = (int*)(ws + off); off += (N1 + 64);
    int*      row_ptr  = (int*)(ws + off); off += (N1 + 64);
    int*      cursor   = (int*)(ws + off); off += (N1 + 64);
    int*      csr      = (int*)(ws + off); off += N_EDGES;
    // total ~40.5 MB

    // LSTM + per-step softmaxes (independent of graph work, tiny)
    lstm_attn_kernel<<<1, 256, 0, stream>>>(query_emb, w_ih, w_hh, b_ih, b_hh,
                                            wl_w, wl_b, r_set,
                                            hiddenWS, attnWS, weightWS, Wf);

    // Build destination-sorted CSR once; reused by all 6 steps (no f32 atomics).
    deg_init_kernel <<<(N1 + 255) / 256, 256, 0, stream>>>(deg);
    deg_count_kernel<<<(N_EDGES + 255) / 256, 256, 0, stream>>>(node_out, deg);
    scan_kernel     <<<1, 256, 0, stream>>>(deg, row_ptr, cursor);
    fill_kernel     <<<(N_EDGES + 255) / 256, 256, 0, stream>>>(node_in, node_out, edge_type,
                                                                cursor, csr);

    const int NB = N1 * 64;
    for (int i = 0; i < N_STEP; ++i) {
        float* out_i = outbufs + (size_t)i * NB;
        inp_kernel   <<<(NB + 255) / 256, 256, 0, stream>>>(
            attnWS + (size_t)i * N_STEP * 64, outbufs, h_set, inp, colsum, i);
        gather_kernel<<<(N1 + 3) / 4, 256, 0, stream>>>(
            row_ptr, csr, inp, weightWS + (size_t)i * NRELX2 * 64, out_i);
        colsum_kernel<<<128, 256, 0, stream>>>(out_i, colsum);
        norm_kernel  <<<(NB + 255) / 256, 256, 0, stream>>>(out_i, colsum);
    }
    score_kernel<<<(2048 + 255) / 256, 256, 0, stream>>>(
        outbufs + (size_t)(N_STEP - 1) * NB, t_index, hr_inverse, lin_w, lin_b, (float*)d_out);
}
